// WindowAttention_41094247088419
// MI455X (gfx1250) — matrix operand order repaired and checked by reading, not compiled, not run on hardware
//
#include <hip/hip_runtime.h>
#include <hip/hip_bf16.h>

typedef __attribute__((ext_vector_type(16))) _Float16 v16h;
typedef __attribute__((ext_vector_type(8)))  _Float16 v8h;
typedef __attribute__((ext_vector_type(8)))  float    v8f;

#define DIM   180
#define HEADS 6
#define HD    30
#define WSZ   8
#define NTOK  64
#define KPAD  192
#define KT    6          // 192 / 32 K-tiles

// d_ws layout (in _Float16 units unless noted):
//   wsQKV : 216 fragment-sets * 6 kc * 512  = 663552 halfs  (3 mats x 6 heads x 2 jn)
//   wsProj: 12 J * 6 kc * 512               = 36864 halfs
//   wsBias: 6*64*64 floats, stored in C/D-fragment order [h][i][j][lane][r]
#define WSQKV_HALFS (216*6*512)
#define WSPROJ_HALFS (12*6*512)

union V16 { v16h v; v8h h[2]; };

// ---------------- prep kernels: pre-swizzle weights into WMMA B-fragment order ----------------
// B operand as packed here (32x16, f16): lane holds column N = lane&15, halfs e=0..15 -> K = e + 16*(lane>>4).
// Fragment element stored at ((frag*6 + kc)*512 + lane*16 + e).

__global__ void prep_qkv_frags(const float* __restrict__ w, _Float16* __restrict__ dst) {
    const int total = WSQKV_HALFS;
    for (int idx = blockIdx.x * blockDim.x + threadIdx.x; idx < total;
         idx += gridDim.x * blockDim.x) {
        int e    = idx & 15;
        int lane = (idx >> 4) & 31;
        int rest = idx >> 9;
        int kc   = rest % 6;
        int f    = rest / 6;          // f = (m*6 + h)*2 + jn
        int m    = f / 12;
        int hjn  = f % 12;
        int h    = hjn >> 1;
        int jn   = hjn & 1;
        int nloc = lane & 15;
        int o    = m * DIM + h * HD + 16 * jn + nloc;   // output row of w_qkv
        int c    = 32 * kc + e + 16 * (lane >> 4);      // input channel (K)
        float v = 0.0f;
        if ((jn == 0 || nloc < 14) && c < DIM) v = w[o * DIM + c];
        dst[idx] = (_Float16)v;
    }
}

__global__ void prep_proj_frags(const float* __restrict__ w, _Float16* __restrict__ dst) {
    const int total = WSPROJ_HALFS;
    for (int idx = blockIdx.x * blockDim.x + threadIdx.x; idx < total;
         idx += gridDim.x * blockDim.x) {
        int e    = idx & 15;
        int lane = (idx >> 4) & 31;
        int rest = idx >> 9;
        int kc   = rest % 6;
        int J    = rest / 6;
        int o    = 16 * J + (lane & 15);
        int c    = 32 * kc + e + 16 * (lane >> 4);
        float v = 0.0f;
        if (o < DIM && c < DIM) v = w[o * DIM + c];
        dst[idx] = (_Float16)v;
    }
}

// Bias in C/D-fragment order: dst[((h*4 + i)*4 + j)*256 + lane*8 + r]
//   = bias[h][n = 16i + r + 8*(lane>>4)][m = 16j + (lane&15)]
__global__ void prep_bias(const float* __restrict__ table, const int* __restrict__ rel,
                          float* __restrict__ dst) {
    const int total = HEADS * NTOK * NTOK;
    for (int idx = blockIdx.x * blockDim.x + threadIdx.x; idx < total;
         idx += gridDim.x * blockDim.x) {
        int r    = idx & 7;
        int lane = (idx >> 3) & 31;
        int j    = (idx >> 8) & 3;
        int i    = (idx >> 10) & 3;
        int h    = idx >> 12;
        int n    = 16 * i + r + 8 * (lane >> 4);
        int m    = 16 * j + (lane & 15);
        dst[idx] = table[rel[n * 64 + m] * HEADS + h];
    }
}

// ---------------- fused window-attention kernel: 1 block = 1 window, 6 waves = 6 heads --------

__global__ __launch_bounds__(192, 1)
void win_attn_kernel(const float* __restrict__ x,
                     const float* __restrict__ b_qkv,
                     const float* __restrict__ b_proj,
                     const _Float16* __restrict__ wq,
                     const _Float16* __restrict__ wp,
                     const float* __restrict__ biasH,
                     float* __restrict__ out) {
    extern __shared__ _Float16 smem[];
    _Float16* xw  = smem;            // [64][192]  input window f16; later reused as attn-out
    _Float16* qs  = smem + 12288;    // [6][64][32] q (pre-scaled), c-pad zeroed
    _Float16* ks  = smem + 24576;    // [6][64][32] k, c-pad zeroed
    _Float16* vsT = smem + 36864;    // [6][32][64] v transposed [d][tok]
    _Float16* sP  = smem + 49152;    // [6][16][64] softmax(P) row-block staging

    const int blk  = blockIdx.x;
    const int b0   = blk >> 10;          // batch
    const int wy   = (blk >> 5) & 31;    // window row
    const int wx   = blk & 31;           // window col
    const int tid  = threadIdx.x;
    const int head = tid >> 5;
    const int lane = tid & 31;
    const int lo   = lane & 15;
    const int hi   = lane >> 4;

    // ---- phase 0: window -> LDS (f16), zero K pad --------------------------------------------
    {
        const size_t base = ((size_t)b0 * DIM) * 65536 + (size_t)(wy * 8) * 256 + wx * 8;
        for (int idx = tid; idx < DIM * WSZ; idx += 192) {
            int c = idx >> 3, iy = idx & 7;
            const float4* src = (const float4*)(x + base + (size_t)c * 65536 + iy * 256);
            float4 a = src[0], bq = src[1];
            _Float16* dst = xw + (iy * 8) * KPAD + c;
            dst[0 * KPAD] = (_Float16)a.x;  dst[1 * KPAD] = (_Float16)a.y;
            dst[2 * KPAD] = (_Float16)a.z;  dst[3 * KPAD] = (_Float16)a.w;
            dst[4 * KPAD] = (_Float16)bq.x; dst[5 * KPAD] = (_Float16)bq.y;
            dst[6 * KPAD] = (_Float16)bq.z; dst[7 * KPAD] = (_Float16)bq.w;
        }
        for (int idx = tid; idx < NTOK * 12; idx += 192) {
            int t = idx / 12, c = DIM + idx % 12;
            xw[t * KPAD + c] = (_Float16)0.0f;
        }
    }
    __syncthreads();

    const float scale = 0.18257418583505536f;  // 30^-0.5

    // ---- phase 1: per-head QKV GEMM (A from LDS, B frags from global/L2) ---------------------
    for (int i = 0; i < 4; ++i) {
        V16 A[KT];
        const _Float16* arow = xw + (i * 16 + lo) * KPAD;
#pragma unroll
        for (int kc = 0; kc < KT; ++kc) {
            int c0 = 32 * kc + 16 * hi;
            A[kc].h[0] = *(const v8h*)(arow + c0);
            A[kc].h[1] = *(const v8h*)(arow + c0 + 8);
        }
#pragma unroll
        for (int m = 0; m < 3; ++m) {
#pragma unroll
            for (int jn = 0; jn < 2; ++jn) {
                const int dd = 16 * jn + lo;
                const bool dvalid = (dd < HD);
                const int o = m * DIM + head * HD + dd;
                float binit = dvalid ? b_qkv[o] : 0.0f;
                v8f acc;
#pragma unroll
                for (int r = 0; r < 8; ++r) acc[r] = binit;
                const int fragBase = (((m * 6 + head) * 2 + jn) * 6) * 512 + lane * 16;
#pragma unroll
                for (int kc = 0; kc < KT; ++kc) {
                    v16h bfr = *(const v16h*)(wq + fragBase + kc * 512);
                    acc = __builtin_amdgcn_wmma_f32_16x16x32_f16(
                        false, A[kc].v, false, bfr, (short)0, acc, false, false);
                }
#pragma unroll
                for (int r = 0; r < 8; ++r) {
                    int t = i * 16 + r + 8 * hi;
                    if (m == 0)
                        qs[(head * 64 + t) * 32 + dd] =
                            dvalid ? (_Float16)(acc[r] * scale) : (_Float16)0.0f;
                    else if (m == 1)
                        ks[(head * 64 + t) * 32 + dd] =
                            dvalid ? (_Float16)acc[r] : (_Float16)0.0f;
                    else
                        vsT[(head * 32 + dd) * 64 + t] =
                            dvalid ? (_Float16)acc[r] : (_Float16)0.0f;
                }
            }
        }
    }
    __syncthreads();

    // zero the proj K-pad columns of the (reused) attn-out buffer
    for (int idx = tid; idx < NTOK * 12; idx += 192) {
        int t = idx / 12, c = DIM + idx % 12;
        xw[t * KPAD + c] = (_Float16)0.0f;
    }

    // ---- phase 2: per-head attention ---------------------------------------------------------
    {
        v16h BK[4], BV0[2], BV1[2];
#pragma unroll
        for (int j = 0; j < 4; ++j)
            BK[j] = *(const v16h*)(ks + (head * 64 + 16 * j + lo) * 32 + 16 * hi);
#pragma unroll
        for (int t = 0; t < 2; ++t) {
            BV0[t] = *(const v16h*)(vsT + (head * 32 + lo) * 64 + 32 * t + 16 * hi);
            BV1[t] = *(const v16h*)(vsT + (head * 32 + 16 + lo) * 64 + 32 * t + 16 * hi);
        }
        const float* bb = biasH + head * 4096 + lane * 8;

        for (int i = 0; i < 4; ++i) {
            V16 Aq;
            const _Float16* qrow = qs + (head * 64 + i * 16 + lo) * 32;
            Aq.h[0] = *(const v8h*)(qrow + 16 * hi);
            Aq.h[1] = *(const v8h*)(qrow + 16 * hi + 8);

            v8f S[4];
#pragma unroll
            for (int j = 0; j < 4; ++j) {
                // bias pre-swizzled into C/D fragment order -> single 32B/lane load
                S[j] = *(const v8f*)(bb + (i * 4 + j) * 256);
                S[j] = __builtin_amdgcn_wmma_f32_16x16x32_f16(
                    false, Aq.v, false, BK[j], (short)0, S[j], false, false);
            }
            // row softmax: row = r + 8*hi (+16*i); columns spread over j and 16 lanes
#pragma unroll
            for (int r = 0; r < 8; ++r) {
                float mx = fmaxf(fmaxf(S[0][r], S[1][r]), fmaxf(S[2][r], S[3][r]));
                mx = fmaxf(mx, __shfl_xor(mx, 1));
                mx = fmaxf(mx, __shfl_xor(mx, 2));
                mx = fmaxf(mx, __shfl_xor(mx, 4));
                mx = fmaxf(mx, __shfl_xor(mx, 8));
                float sum = 0.0f;
#pragma unroll
                for (int j = 0; j < 4; ++j) {
                    float p = __expf(S[j][r] - mx);
                    S[j][r] = p;
                    sum += p;
                }
                sum += __shfl_xor(sum, 1);
                sum += __shfl_xor(sum, 2);
                sum += __shfl_xor(sum, 4);
                sum += __shfl_xor(sum, 8);
                float inv = 1.0f / sum;
#pragma unroll
                for (int j = 0; j < 4; ++j)
                    sP[(head * 16 + r + 8 * hi) * 64 + 16 * j + lo] =
                        (_Float16)(S[j][r] * inv);
            }
            // restage P (C/D layout) -> A-fragment layout via LDS, then O = P @ V
            V16 Ap[2];
            const _Float16* prow = sP + (head * 16 + lo) * 64;
#pragma unroll
            for (int t = 0; t < 2; ++t) {
                Ap[t].h[0] = *(const v8h*)(prow + 32 * t + 16 * hi);
                Ap[t].h[1] = *(const v8h*)(prow + 32 * t + 16 * hi + 8);
            }
            v8f O0 = {}, O1 = {};
            O0 = __builtin_amdgcn_wmma_f32_16x16x32_f16(false, Ap[0].v, false, BV0[0], (short)0, O0, false, false);
            O0 = __builtin_amdgcn_wmma_f32_16x16x32_f16(false, Ap[1].v, false, BV0[1], (short)0, O0, false, false);
            O1 = __builtin_amdgcn_wmma_f32_16x16x32_f16(false, Ap[0].v, false, BV1[0], (short)0, O1, false, false);
            O1 = __builtin_amdgcn_wmma_f32_16x16x32_f16(false, Ap[1].v, false, BV1[1], (short)0, O1, false, false);
#pragma unroll
            for (int r = 0; r < 8; ++r) {
                int t = i * 16 + r + 8 * hi;
                xw[t * KPAD + head * HD + lo] = (_Float16)O0[r];
                if (lo < 14) xw[t * KPAD + head * HD + 16 + lo] = (_Float16)O1[r];
            }
        }
    }
    __syncthreads();

    // ---- phase 3: proj GEMM + window reverse (vectorized b128 stores) ------------------------
    for (int i = 0; i < 4; ++i) {
        V16 A[KT];
        const _Float16* arow = xw + (i * 16 + lo) * KPAD;
#pragma unroll
        for (int kc = 0; kc < KT; ++kc) {
            int c0 = 32 * kc + 16 * hi;
            A[kc].h[0] = *(const v8h*)(arow + c0);
            A[kc].h[1] = *(const v8h*)(arow + c0 + 8);
        }
#pragma unroll
        for (int jj = 0; jj < 2; ++jj) {
            const int J = head * 2 + jj;
            const int o = 16 * J + lo;
            float binit = (o < DIM) ? b_proj[o] : 0.0f;
            v8f acc;
#pragma unroll
            for (int r = 0; r < 8; ++r) acc[r] = binit;
#pragma unroll
            for (int kc = 0; kc < KT; ++kc) {
                v16h bfr = *(const v16h*)(wp + (J * 6 + kc) * 512 + lane * 16);
                acc = __builtin_amdgcn_wmma_f32_16x16x32_f16(
                    false, A[kc].v, false, bfr, (short)0, acc, false, false);
            }
            if (o < DIM) {
                // lane owns output channel o; its 8 tile elements are consecutive in x:
                // t = 16i + r + 8hi -> iy = 2i + hi (const), ix = r
                float* op = out + ((size_t)b0 * DIM + o) * 65536 +
                            (size_t)(wy * 8 + 2 * i + hi) * 256 + wx * 8;
                float4 s0 = make_float4(acc[0], acc[1], acc[2], acc[3]);
                float4 s1 = make_float4(acc[4], acc[5], acc[6], acc[7]);
                *(float4*)(op)     = s0;
                *(float4*)(op + 4) = s1;
            }
        }
    }
}

// ---------------- launch ----------------------------------------------------------------------
extern "C" void kernel_launch(void* const* d_in, const int* in_sizes, int n_in,
                              void* d_out, int out_size, void* d_ws, size_t ws_size,
                              hipStream_t stream) {
    const float* x          = (const float*)d_in[0];
    const float* w_qkv      = (const float*)d_in[1];
    const float* b_qkv      = (const float*)d_in[2];
    const float* w_proj     = (const float*)d_in[3];
    const float* b_proj     = (const float*)d_in[4];
    const float* bias_table = (const float*)d_in[5];
    const int*   rel_idx    = (const int*)d_in[6];
    float* out = (float*)d_out;

    _Float16* wsQKV  = (_Float16*)d_ws;
    _Float16* wsProj = wsQKV + WSQKV_HALFS;
    float*    wsBias = (float*)(wsProj + WSPROJ_HALFS);

    prep_qkv_frags<<<648, 256, 0, stream>>>(w_qkv, wsQKV);
    prep_proj_frags<<<144, 256, 0, stream>>>(w_proj, wsProj);
    prep_bias<<<96, 256, 0, stream>>>(bias_table, rel_idx, wsBias);

    const int smemBytes = 110592;  // 108 KB dynamic LDS
    hipFuncSetAttribute(reinterpret_cast<const void*>(win_attn_kernel),
                        hipFuncAttributeMaxDynamicSharedMemorySize, smemBytes);
    win_attn_kernel<<<8192, 192, smemBytes, stream>>>(x, b_qkv, b_proj, wsQKV, wsProj,
                                                      wsBias, out);
}
